// ResponseBaselineProt_54142357733864
// MI455X (gfx1250) — matrix operand order repaired and checked by reading, not compiled, not run on hardware
//
#include <hip/hip_runtime.h>
#include <hip/hip_bf16.h>

typedef _Float16 v16h __attribute__((ext_vector_type(16)));
typedef float    v8f  __attribute__((ext_vector_type(8)));

#define N_PROT  19000
#define N_DRUGS 4200
#define BATCH   8192
#define D0      1024
#define D1      512
#define D2      128
#define MROWS   (2 * BATCH)          // 16384 sample-slot rows
#define KT0     (D0 / 32)            // 32 K-tiles feeding gemm1
#define KT1     (D1 / 32)            // 16 K-tiles feeding gemm2
#define NT1     (D1 / 16)            // 32 N-tiles of W1
#define NT2     (D2 / 16)            // 8  N-tiles of W2

// ---------------------------------------------------------------------------
// Kernel A: P[d][j] = sum_{p : table[d][p] != 0} W0[p][j]   (sparse, ~38 nnz/row)
// One block per drug. Table row staged through LDS in 256-float chunks; W0 rows
// touched only when the entry is set.
// ---------------------------------------------------------------------------
__global__ __launch_bounds__(256) void build_P(const float* __restrict__ table,
                                               const float* __restrict__ W0,
                                               float* __restrict__ P) {
  const int d = blockIdx.x;
  const int t = threadIdx.x;
  __shared__ float tv[256];
  float acc0 = 0.f, acc1 = 0.f, acc2 = 0.f, acc3 = 0.f;
  const float* row = table + (size_t)d * N_PROT;
  for (int p0 = 0; p0 < N_PROT; p0 += 256) {
    const int p = p0 + t;
    tv[t] = (p < N_PROT) ? row[p] : 0.f;
    __syncthreads();
    const int lim = (N_PROT - p0) < 256 ? (N_PROT - p0) : 256;
    for (int q = 0; q < lim; ++q) {
      if (tv[q] != 0.f) {
        const float* w = W0 + (size_t)(p0 + q) * D0 + t;
        acc0 += w[0]; acc1 += w[256]; acc2 += w[512]; acc3 += w[768];
      }
    }
    __syncthreads();
  }
  float* out = P + (size_t)d * D0 + t;
  out[0] = acc0; out[256] = acc1; out[512] = acc2; out[768] = acc3;
}

// ---------------------------------------------------------------------------
// B-operand swizzle (32K x 16N f16 tile, 512 halves contiguous per tile):
//   lane = n + 16*g with g=(K>>3)&1 ; half h = (K&7) + (K>=16 ? 8 : 0)
// ---------------------------------------------------------------------------
__global__ void pack_B(const float* __restrict__ W, _Float16* __restrict__ Wt,
                       int K, int N) {
  const size_t idx = (size_t)blockIdx.x * 256 + threadIdx.x;
  if (idx >= (size_t)K * N) return;
  const int k = (int)(idx / N);
  const int n = (int)(idx % N);
  const int kt = k >> 5, kl = k & 31;
  const int nt = n >> 4, nl = n & 15;
  const int g = (kl >> 3) & 1;
  const int h = (kl & 7) + ((kl >= 16) ? 8 : 0);
  const int lane = nl + 16 * g;
  const size_t ntiles = (size_t)(N >> 4);
  Wt[((size_t)kt * ntiles + nt) * 512 + (size_t)lane * 16 + h] =
      (_Float16)W[(size_t)k * N + n];
}

// ---------------------------------------------------------------------------
// A-operand swizzle (16M x 32K f16 tile per ISA 7.12.2):
//   lane = m + 16*g with g=(K>>3)&1 ; half h = (K&7) + (K>=16 ? 8 : 0)
// ---------------------------------------------------------------------------
__device__ __forceinline__ size_t a_swizzle(int m, int K) {
  const int g = (K >> 3) & 1;
  const int h = (K & 7) + ((K >= 16) ? 8 : 0);
  return (size_t)(m + 16 * g) * 16 + h;
}

// Kernel B: h0 = relu(P[drug] + conc * W0[19000] + b0), written in A-layout f16.
// Rows 0..8191 = slot 0, rows 8192..16383 = slot 1.
__global__ __launch_bounds__(256) void build_H0(const float* __restrict__ P,
                                                const float* __restrict__ W0last,
                                                const float* __restrict__ b0,
                                                const int*   __restrict__ pairs,
                                                const float* __restrict__ conc,
                                                _Float16* __restrict__ H0t) {
  const size_t idx = (size_t)blockIdx.x * 256 + threadIdx.x; // 16384*1024
  const int j = (int)(idx & (D0 - 1));
  const int r = (int)(idx >> 10);
  const int slot = r >> 13;
  const int i = r & (BATCH - 1);
  const int d = pairs[i * 2 + slot];
  const float c = conc[i * 3 + 1 + slot];
  float v = P[(size_t)d * D0 + j] + c * W0last[j] + b0[j];
  v = v > 0.f ? v : 0.f;
  const int rt = r >> 4, m = r & 15;
  const int kt = j >> 5, K = j & 31;
  H0t[((size_t)rt * KT0 + kt) * 512 + a_swizzle(m, K)] = (_Float16)v;
}

// ---------------------------------------------------------------------------
// Kernel C: H1 = relu(H0 @ W1 + b1)   (16384x1024 @ 1024x512), f16 WMMA, f32 acc.
// Block covers 64M x 512N: 8 waves, each wave owns a 64M x 64N tile
// (4 M-tiles x 4 N-tiles -> 16 v8f accumulators). Each B tile is reused by
// 4 WMMAs, cutting L2 B-traffic 4x vs a 16-row block. K-loop unroll capped to
// bound VGPR pressure (~200) so all 8 waves co-reside.
// ---------------------------------------------------------------------------
__global__ __launch_bounds__(256) void gemm1(const _Float16* __restrict__ At,
                                             const _Float16* __restrict__ Bt,
                                             const float*    __restrict__ bias,
                                             _Float16*       __restrict__ H1t) {
  const int bm   = blockIdx.x;        // 256 blocks, each 4 M-tiles
  const int wave = threadIdx.x >> 5;  // 0..7 -> N-tiles 4w..4w+3
  const int lane = threadIdx.x & 31;
  v8f acc[4][4] = {};
#pragma unroll 2
  for (int kt = 0; kt < KT0; ++kt) {
    v16h a[4];
#pragma unroll
    for (int mt = 0; mt < 4; ++mt)
      a[mt] = *(const v16h*)(At + ((size_t)(bm * 4 + mt) * KT0 + kt) * 512 +
                             (size_t)lane * 16);
    const _Float16* bbase =
        Bt + ((size_t)kt * NT1 + wave * 4) * 512 + (size_t)lane * 16;
#pragma unroll
    for (int q = 0; q < 4; ++q) {
      const v16h b = *(const v16h*)(bbase + (size_t)q * 512);
#pragma unroll
      for (int mt = 0; mt < 4; ++mt)
        acc[mt][q] = __builtin_amdgcn_wmma_f32_16x16x32_f16(
            false, a[mt], false, b, (short)0, acc[mt][q], false, false);
    }
  }
#pragma unroll
  for (int mt = 0; mt < 4; ++mt) {
    const int rt = bm * 4 + mt;
#pragma unroll
    for (int q = 0; q < 4; ++q) {
      const int nglob = (wave * 4 + q) * 16 + (lane & 15);
      const float bn = bias[nglob];
      const int kt2 = nglob >> 5, K = nglob & 31;
      const size_t tbase = ((size_t)rt * KT1 + kt2) * 512;
#pragma unroll
      for (int v = 0; v < 8; ++v) {
        const int m = v + ((lane >> 4) << 3);   // C layout: M = vgpr + 8*(lane/16)
        float val = acc[mt][q][v] + bn;
        val = val > 0.f ? val : 0.f;
        H1t[tbase + a_swizzle(m, K)] = (_Float16)val;
      }
    }
  }
}

// Kernel D: Z = H1 @ W2 + b2   (16384x512 @ 512x128), f32 row-major output.
__global__ __launch_bounds__(256) void gemm2(const _Float16* __restrict__ At,
                                             const _Float16* __restrict__ Bt,
                                             const float*    __restrict__ bias,
                                             float*          __restrict__ Z) {
  const int rt   = blockIdx.x;        // 1024 row-tiles
  const int wave = threadIdx.x >> 5;  // 8 waves == 8 N-tiles
  const int lane = threadIdx.x & 31;
  v8f acc = {};
  for (int kt = 0; kt < KT1; ++kt) {
    const v16h a = *(const v16h*)(At + ((size_t)rt * KT1 + kt) * 512 +
                                  (size_t)lane * 16);
    const v16h b = *(const v16h*)(Bt + ((size_t)kt * NT2 + wave) * 512 +
                                  (size_t)lane * 16);
    acc = __builtin_amdgcn_wmma_f32_16x16x32_f16(
        false, a, false, b, (short)0, acc, false, false);
  }
  const int n = wave * 16 + (lane & 15);
  const float bn = bias[n];
#pragma unroll
  for (int v = 0; v < 8; ++v) {
    const int m = v + ((lane >> 4) << 3);
    Z[(size_t)(rt * 16 + m) * D2 + n] = acc[v] + bn;
  }
}

// Kernel E: out[i] = dot(Z[i], Z[8192+i])
__global__ __launch_bounds__(256) void dot_z(const float* __restrict__ Z,
                                             float* __restrict__ out) {
  const int i = blockIdx.x * 256 + threadIdx.x;
  const float4* z1 = (const float4*)(Z + (size_t)i * D2);
  const float4* z2 = (const float4*)(Z + (size_t)(i + BATCH) * D2);
  float s = 0.f;
#pragma unroll
  for (int q = 0; q < D2 / 4; ++q) {
    const float4 a = z1[q], b = z2[q];
    s += a.x * b.x + a.y * b.y + a.z * b.z + a.w * b.w;
  }
  out[i] = s;
}

// ---------------------------------------------------------------------------
extern "C" void kernel_launch(void* const* d_in, const int* in_sizes, int n_in,
                              void* d_out, int out_size, void* d_ws, size_t ws_size,
                              hipStream_t stream) {
  const float* table = (const float*)d_in[0];
  const int*   pairs = (const int*)  d_in[1];
  const float* conc  = (const float*)d_in[2];
  const float* W0    = (const float*)d_in[3];
  const float* b0    = (const float*)d_in[4];
  const float* W1    = (const float*)d_in[5];
  const float* b1    = (const float*)d_in[6];
  const float* W2    = (const float*)d_in[7];
  const float* b2    = (const float*)d_in[8];
  float* out = (float*)d_out;

  // Workspace carve-out (1KB-aligned regions).
  char* ws = (char*)d_ws;
  size_t off = 0;
  auto carve = [&](size_t bytes) {
    char* p = ws + off;
    off += (bytes + 1023) & ~(size_t)1023;
    return p;
  };
  float*    P    = (float*)   carve((size_t)N_DRUGS * D0 * 4);   // 17.2 MB
  _Float16* W1t  = (_Float16*)carve((size_t)D0 * D1 * 2);        //  1.0 MB
  _Float16* W2t  = (_Float16*)carve((size_t)D1 * D2 * 2);        //  0.1 MB
  _Float16* H0t  = (_Float16*)carve((size_t)MROWS * D0 * 2);     // 33.6 MB
  _Float16* H1t  = (_Float16*)carve((size_t)MROWS * D1 * 2);     // 16.8 MB
  float*    Z    = (float*)   carve((size_t)MROWS * D2 * 4);     //  8.4 MB
  (void)ws_size; (void)in_sizes; (void)n_in; (void)out_size;

  build_P<<<N_DRUGS, 256, 0, stream>>>(table, W0, P);

  pack_B<<<((size_t)D0 * D1 + 255) / 256, 256, 0, stream>>>(W1, W1t, D0, D1);
  pack_B<<<((size_t)D1 * D2 + 255) / 256, 256, 0, stream>>>(W2, W2t, D1, D2);

  build_H0<<<((size_t)MROWS * D0) / 256, 256, 0, stream>>>(
      P, W0 + (size_t)N_PROT * D0, b0, pairs, conc, H0t);

  gemm1<<<MROWS / 64, 256, 0, stream>>>(H0t, W1t, b1, H1t);
  gemm2<<<MROWS / 16, 256, 0, stream>>>(H1t, W2t, b2, Z);
  dot_z<<<BATCH / 256, 256, 0, stream>>>(Z, out);
}
